// Net_76450417869267
// MI455X (gfx1250) — matrix operand order repaired and checked by reading, not compiled, not run on hardware
//
#include <hip/hip_runtime.h>
#include <math.h>

// ---------------- types for CDNA5 WMMA (wave32, bf16 in / f32 acc) ----------
typedef __bf16 bf16;
typedef bf16  v8bf  __attribute__((ext_vector_type(8)));
typedef bf16  v16bf __attribute__((ext_vector_type(16)));
typedef float v8f   __attribute__((ext_vector_type(8)));
typedef int   v4i   __attribute__((ext_vector_type(4)));

#define T_STEPS 96
#define BATCH   256
#define HID     512
#define GATES   2048          // 4*HID
#define DIN_PAD 96            // 1 + 20 + 64 = 85, padded to 96
#define K0CAT   608           // layer0: 96 + 512   (19 k-tiles of 32)
#define K1CAT   1024          // layer1: 512 + 512  (32 k-tiles of 32)
#define LOG2PI  1.8378770664093453f

// LDS staging: per k-tile stage A (128 rows x 64B) and B (64 rows x 64B),
// rows padded to 80B to spread banks. Double buffered.
#define ROWB    80
#define B_OFF   (128 * ROWB)             // 10240
#define BUFSZ   (B_OFF + 64 * ROWB)      // 15360 ; 2*BUFSZ = 30720 B shared

// -------- CDNA5 async global->LDS path (guarded; falls back to sync copy) ---
#if defined(__HIP_DEVICE_COMPILE__) && defined(__has_builtin)
#if __has_builtin(__builtin_amdgcn_global_load_async_to_lds_b128)
#define HAVE_ASYNC 1
#endif
#endif

#ifdef HAVE_ASYNC
typedef __attribute__((address_space(1))) v4i* gv4p;   // global int4*
typedef __attribute__((address_space(3))) v4i* lv4p;   // LDS int4*
__device__ __forceinline__ void cp16(const bf16* g, char* l) {
    __builtin_amdgcn_global_load_async_to_lds_b128((gv4p)g, (lv4p)l, 0, 0);
}
#if __has_builtin(__builtin_amdgcn_s_wait_asynccnt)
#define ASYNC_WAIT(n) __builtin_amdgcn_s_wait_asynccnt(n)
#else
#define ASYNC_WAIT(n) asm volatile("s_wait_asynccnt %0" :: "i"(n) : "memory")
#endif
#else
__device__ __forceinline__ void cp16(const bf16* g, char* l) {
    *(v8bf*)l = *(const v8bf*)g;          // sync fallback: vmem load + ds store
}
#define ASYNC_WAIT(n) do {} while (0)
#endif

// ---------------- helpers ---------------------------------------------------
__device__ __forceinline__ bf16 f2bf(float f) {
    union { float f; unsigned u; } in; in.f = f;
    unsigned r = in.u + 0x7FFFu + ((in.u >> 16) & 1u);   // round-nearest-even
    union { unsigned short u; bf16 b; } out; out.u = (unsigned short)(r >> 16);
    return out.b;
}
__device__ __forceinline__ float sigm(float x) { return 1.0f / (1.0f + expf(-x)); }

// ---------------- setup kernels (run every call; deterministic) -------------
// fp32 (rows x srcK) -> bf16 block at dst[r*dstStride + colOff + k], k<dstK, pad 0
__global__ void conv_w(const float* __restrict__ src, bf16* __restrict__ dst,
                       int rows, int srcK, int dstK, int dstStride, int colOff) {
    int i = blockIdx.x * 256 + threadIdx.x;
    if (i >= rows * dstK) return;
    int k = i % dstK; int r = i / dstK;
    float v = (k < srcK) ? src[r * srcK + k] : 0.0f;
    dst[r * dstStride + colOff + k] = f2bf(v);
}

__global__ void bias_comb(const float* a, const float* b, float* o) {
    int i = blockIdx.x * 256 + threadIdx.x;
    if (i < GATES) o[i] = a[i] + b[i];
}

// init h/c state in d_out, bf16 h-shadows inside concatenated activations
__global__ void init_state(const float* __restrict__ hidden, const float* __restrict__ cell,
                           float* __restrict__ out, bf16* __restrict__ Acat0,
                           bf16* __restrict__ Acat1, float* __restrict__ mu_prev) {
    int i = blockIdx.x * 256 + threadIdx.x;           // 0 .. 2*B*H-1
    if (i < 2 * BATCH * HID) {
        out[512 + i]                   = hidden[i];   // h state
        out[512 + 2 * BATCH * HID + i] = cell[i];     // c state
        bf16 hv = f2bf(hidden[i]);
        int l = i >> 17, r = i & (BATCH * HID - 1);
        int b = r >> 9, j = r & (HID - 1);
        if (l == 0) { Acat0[b * K0CAT + DIN_PAD + j] = hv; Acat1[b * K1CAT + j] = hv; }
        else        { Acat1[b * K1CAT + HID + j] = hv; }
    }
    if (i < BATCH) mu_prev[i] = 0.0f;
    if (i == 0)    out[512 + 4 * BATCH * HID] = 0.0f; // loss
}

// ---------------- per-step input build (carries mu recurrence) --------------
// fills Acat0[b, 0..95] = [ z0 | cov(20) | emb(64) | pad(11) ]
__global__ void build_x(const float* __restrict__ train, const int* __restrict__ idx,
                        const float* __restrict__ embed, const float* __restrict__ mu_prev,
                        bf16* __restrict__ Acat0, int t) {
    int i = blockIdx.x * 256 + threadIdx.x;           // 0 .. 256*96-1
    if (i >= BATCH * DIN_PAD) return;
    int b = i / DIN_PAD, k = i % DIN_PAD;
    float v;
    if (k == 0) {
        float x0 = train[(t * BATCH + b) * 21];
        v = (t > 0 && x0 == 0.0f) ? mu_prev[b] : x0;
    } else if (k <= 20) {
        v = train[(t * BATCH + b) * 21 + k];
    } else if (k <= 84) {
        v = embed[idx[b] * 64 + (k - 21)];
    } else v = 0.0f;
    Acat0[b * K0CAT + k] = f2bf(v);
}

// ---------------- WMMA gates GEMM: C(256x2048) = A(256xK) @ W(2048xK)^T -----
// Block = 8 waves = one 128x64 macro-tile. Single branch-free k-loop; A/B
// k-tiles staged in LDS via async global->LDS B128, double buffered.
__global__ void __launch_bounds__(256)
gates_gemm(const bf16* __restrict__ A, const bf16* __restrict__ W, int kt,
           float* __restrict__ C) {
    __shared__ __align__(16) char smem[2 * BUFSZ];
    const int tid  = threadIdx.x;
    const int lane = tid & 31;
    const int w    = tid >> 5;                  // wave 0..7
    const int mBase = (blockIdx.x & 1) * 128;   // m half
    const int n0    = (blockIdx.x >> 1) * 64;   // 32 n-groups of 64 columns
    const int m0    = mBase + w * 16;
    const int ld    = kt * 32;                  // row stride == K (concatenated)

    // ---- loop-invariant staging addresses (3 async b128 per thread/tile) ----
    const int srow = tid >> 2, spart = tid & 3;
    const bf16* gA0 = A + (mBase + srow) * ld + spart * 8;        // A rows 0..63
    const bf16* gA1 = A + (mBase + 64 + srow) * ld + spart * 8;   // A rows 64..127
    const bf16* gB  = W + (n0 + srow) * ld + spart * 8;           // B rows 0..63
    const int lA0 = srow * ROWB + spart * 16;
    const int lA1 = (64 + srow) * ROWB + spart * 16;
    const int lB  = B_OFF + srow * ROWB + spart * 16;

    // ---- loop-invariant compute addresses ----
    const int mrow = lane & 15;                 // A row in tile / B column
    const int half = lane >> 4;                 // k-half select
    const int apoff = (w * 16 + mrow) * ROWB;
    const int bpoff = B_OFF + mrow * ROWB + half * 16;

    v8f acc[4];
#pragma unroll
    for (int n = 0; n < 4; ++n)
#pragma unroll
        for (int j = 0; j < 8; ++j) acc[n][j] = 0.0f;

    // prologue: stage k-tile 0
    cp16(gA0, smem + lA0); cp16(gA1, smem + lA1); cp16(gB, smem + lB);

    for (int k = 0; k < kt; ++k) {
        if (k + 1 < kt) {                       // stage k+1 into other buffer
            char* nb = smem + ((k + 1) & 1) * BUFSZ;
            int ko = (k + 1) * 32;
            cp16(gA0 + ko, nb + lA0); cp16(gA1 + ko, nb + lA1); cp16(gB + ko, nb + lB);
            ASYNC_WAIT(3);                      // tile k landed; k+1 in flight
        } else {
            ASYNC_WAIT(0);
        }
        __syncthreads();

        const char* buf = smem + (k & 1) * BUFSZ;
        // issue all 10 ds_load_b128 first, then 4 wmma (partial dscnt waits)
        const char* ap = buf + apoff;
        v8bf alo = *(const v8bf*)(ap + half * 16);
        v8bf ahi = *(const v8bf*)(ap + 32 + half * 16);
        const char* bp = buf + bpoff;
        v8bf b0l = *(const v8bf*)(bp);
        v8bf b0h = *(const v8bf*)(bp + 32);
        v8bf b1l = *(const v8bf*)(bp + 16 * ROWB);
        v8bf b1h = *(const v8bf*)(bp + 16 * ROWB + 32);
        v8bf b2l = *(const v8bf*)(bp + 32 * ROWB);
        v8bf b2h = *(const v8bf*)(bp + 32 * ROWB + 32);
        v8bf b3l = *(const v8bf*)(bp + 48 * ROWB);
        v8bf b3h = *(const v8bf*)(bp + 48 * ROWB + 32);

        v16bf a, bm0, bm1, bm2, bm3;
#pragma unroll
        for (int i = 0; i < 8; ++i) {
            a[i] = alo[i];   a[i + 8] = ahi[i];
            bm0[i] = b0l[i]; bm0[i + 8] = b0h[i];
            bm1[i] = b1l[i]; bm1[i + 8] = b1h[i];
            bm2[i] = b2l[i]; bm2[i + 8] = b2h[i];
            bm3[i] = b3l[i]; bm3[i + 8] = b3h[i];
        }
        acc[0] = __builtin_amdgcn_wmma_f32_16x16x32_bf16(false, a, false, bm0, (short)0, acc[0], false, false);
        acc[1] = __builtin_amdgcn_wmma_f32_16x16x32_bf16(false, a, false, bm1, (short)0, acc[1], false, false);
        acc[2] = __builtin_amdgcn_wmma_f32_16x16x32_bf16(false, a, false, bm2, (short)0, acc[2], false, false);
        acc[3] = __builtin_amdgcn_wmma_f32_16x16x32_bf16(false, a, false, bm3, (short)0, acc[3], false, false);
        __syncthreads();                        // done reading before overwrite
    }

    // C/D layout: VGPR j -> M=j (lanes 0-15) / M=j+8 (lanes 16-31), N = lane&15
    const int col = lane & 15, rbase = (lane >> 4) * 8;
#pragma unroll
    for (int n = 0; n < 4; ++n)
#pragma unroll
        for (int j = 0; j < 8; ++j)
            C[(m0 + rbase + j) * GATES + n0 + n * 16 + col] = acc[n][j];
}

// ---------------- pointwise LSTM cell ---------------------------------------
// writes f32 h/c state and bf16 h-shadow(s) into concatenated activation bufs
__global__ void lstm_cell(const float* __restrict__ gates, const float* __restrict__ bias,
                          float* __restrict__ h_out, float* __restrict__ c_state,
                          bf16* __restrict__ hb1, int ld1,
                          bf16* __restrict__ hb2, int ld2) {
    int idx = blockIdx.x * 256 + threadIdx.x;       // 0 .. B*H-1
    int b = idx >> 9, j = idx & (HID - 1);
    const float* g = gates + b * GATES;
    float gi = g[j]           + bias[j];
    float gf = g[j + HID]     + bias[j + HID];
    float gg = g[j + 2 * HID] + bias[j + 2 * HID];
    float go = g[j + 3 * HID] + bias[j + 3 * HID];
    float c  = sigm(gf) * c_state[idx] + sigm(gi) * tanhf(gg);
    float h  = sigm(go) * tanhf(c);
    c_state[idx] = c;
    h_out[idx]   = h;
    bf16 hv = f2bf(h);
    hb1[b * ld1 + j] = hv;
    if (hb2) hb2[b * ld2 + j] = hv;
}

// ---------------- projection + masked Gaussian NLL (one block) --------------
__global__ void proj_loss(const float* __restrict__ h0, const float* __restrict__ h1,
                          const float* __restrict__ Wmu, const float* __restrict__ bmu,
                          const float* __restrict__ Wps, const float* __restrict__ bps,
                          const float* __restrict__ labels_t,
                          float* __restrict__ mu_prev,
                          float* __restrict__ out_mu, float* __restrict__ out_sigma,
                          float* __restrict__ out_loss) {
    int b = threadIdx.x;                             // 256 threads
    const float* r0 = h0 + b * HID;
    const float* r1 = h1 + b * HID;
    float m = 0.0f, p = 0.0f;
    for (int j = 0; j < HID; ++j) {                  // hp[b,2j]=h0, hp[b,2j+1]=h1
        float a = r0[j], c = r1[j];
        m += a * Wmu[2 * j] + c * Wmu[2 * j + 1];
        p += a * Wps[2 * j] + c * Wps[2 * j + 1];
    }
    float mu = m + bmu[0];
    float ps = p + bps[0];
    float sigma = (ps > 20.0f) ? ps : log1pf(expf(ps));   // softplus
    float lab = labels_t[b];
    bool  msk = (lab != 0.0f);
    float d = (lab - mu) / sigma;
    float logp = -0.5f * d * d - logf(sigma) - 0.5f * LOG2PI;

    __shared__ float s_sum[256];
    __shared__ int   s_cnt[256];
    s_sum[b] = msk ? logp : 0.0f;
    s_cnt[b] = msk ? 1 : 0;
    __syncthreads();
    for (int st = 128; st > 0; st >>= 1) {
        if (b < st) { s_sum[b] += s_sum[b + st]; s_cnt[b] += s_cnt[b + st]; }
        __syncthreads();
    }
    mu_prev[b]   = mu;
    out_mu[b]    = mu;       // overwritten each step; last step persists
    out_sigma[b] = sigma;
    if (b == 0) {
        int n = (s_cnt[0] > 1) ? s_cnt[0] : 1;
        out_loss[0] += -s_sum[0] / (float)n;
    }
}

// ---------------- launch ----------------------------------------------------
extern "C" void kernel_launch(void* const* d_in, const int* in_sizes, int n_in,
                              void* d_out, int out_size, void* d_ws, size_t ws_size,
                              hipStream_t stream) {
    const float* train  = (const float*)d_in[0];
    const int*   idx    = (const int*)  d_in[1];
    const float* hidden = (const float*)d_in[2];
    const float* cell   = (const float*)d_in[3];
    const float* labels = (const float*)d_in[4];
    const float* embed  = (const float*)d_in[5];
    const float* W_ih0  = (const float*)d_in[6];
    const float* W_hh0  = (const float*)d_in[7];
    const float* b_ih0  = (const float*)d_in[8];
    const float* b_hh0  = (const float*)d_in[9];
    const float* W_ih1  = (const float*)d_in[10];
    const float* W_hh1  = (const float*)d_in[11];
    const float* b_ih1  = (const float*)d_in[12];
    const float* b_hh1  = (const float*)d_in[13];
    const float* W_mu   = (const float*)d_in[14];
    const float* b_mu   = (const float*)d_in[15];
    const float* W_ps   = (const float*)d_in[16];
    const float* b_ps   = (const float*)d_in[17];

    char* ws = (char*)d_ws;
    bf16*  Wcat0  = (bf16*)(ws + 0);          // [Wih0|Whh0] 2048x608 bf16 = 2490368 B
    bf16*  Wcat1  = (bf16*)(ws + 2490368);    // [Wih1|Whh1] 2048x1024 bf16 = 4194304 B
    bf16*  Acat0  = (bf16*)(ws + 6684672);    // [x|h0] 256x608 bf16 = 311296 B
    bf16*  Acat1  = (bf16*)(ws + 6995968);    // [h0|h1] 256x1024 bf16 = 524288 B
    float* gatesB = (float*)(ws + 7520256);   // 256*2048 f32 = 2097152 B (reused)
    float* b0comb = (float*)(ws + 9617408);   // 2048*4
    float* b1comb = (float*)(ws + 9625600);
    float* muprev = (float*)(ws + 9633792);   // 256*4

    float* out     = (float*)d_out;
    float* out_mu  = out;                               // [0,256)
    float* out_sig = out + 256;                         // [256,512)
    float* h0_st   = out + 512;                         // h layer0 (B,H)
    float* h1_st   = out + 512 + BATCH * HID;           // h layer1
    float* c0_st   = out + 512 + 2 * BATCH * HID;       // c layer0
    float* c1_st   = out + 512 + 3 * BATCH * HID;       // c layer1
    float* outloss = out + 512 + 4 * BATCH * HID;       // scalar

    // ---- one-time setup (every call; pure function of inputs) ----
    conv_w<<<(GATES * DIN_PAD + 255) / 256, 256, 0, stream>>>(W_ih0, Wcat0, GATES, 85, DIN_PAD, K0CAT, 0);
    conv_w<<<(GATES * HID + 255) / 256, 256, 0, stream>>>(W_hh0, Wcat0, GATES, HID, HID, K0CAT, DIN_PAD);
    conv_w<<<(GATES * HID + 255) / 256, 256, 0, stream>>>(W_ih1, Wcat1, GATES, HID, HID, K1CAT, 0);
    conv_w<<<(GATES * HID + 255) / 256, 256, 0, stream>>>(W_hh1, Wcat1, GATES, HID, HID, K1CAT, HID);
    bias_comb<<<GATES / 256, 256, 0, stream>>>(b_ih0, b_hh0, b0comb);
    bias_comb<<<GATES / 256, 256, 0, stream>>>(b_ih1, b_hh1, b1comb);
    init_state<<<(2 * BATCH * HID) / 256, 256, 0, stream>>>(hidden, cell, out, Acat0, Acat1, muprev);

    // ---- sequential recurrence: 6 chained kernels per timestep ----
    for (int t = 0; t < T_STEPS; ++t) {
        build_x<<<(BATCH * DIN_PAD + 255) / 256, 256, 0, stream>>>(train, idx, embed, muprev, Acat0, t);
        // layer 0: gates = [x|h0] @ [Wih0|Whh0]^T   (K = 608)
        gates_gemm<<<64, 256, 0, stream>>>(Acat0, Wcat0, K0CAT / 32, gatesB);
        lstm_cell<<<(BATCH * HID) / 256, 256, 0, stream>>>(gatesB, b0comb, h0_st, c0_st,
                                                           Acat0 + DIN_PAD, K0CAT, Acat1, K1CAT);
        // layer 1: gates = [h0|h1] @ [Wih1|Whh1]^T  (K = 1024)
        gates_gemm<<<64, 256, 0, stream>>>(Acat1, Wcat1, K1CAT / 32, gatesB);
        lstm_cell<<<(BATCH * HID) / 256, 256, 0, stream>>>(gatesB, b1comb, h1_st, c1_st,
                                                           Acat1 + HID, K1CAT, (bf16*)nullptr, 0);
        proj_loss<<<1, 256, 0, stream>>>(h0_st, h1_st, W_mu, b_mu, W_ps, b_ps,
                                         labels + t * BATCH, muprev, out_mu, out_sig, outloss);
    }
}
